// mLSTM_49349174231641
// MI455X (gfx1250) — matrix operand order repaired and checked by reading, not compiled, not run on hardware
//
#include <hip/hip_runtime.h>
#include <hip/hip_bf16.h>
#include <math.h>

typedef __attribute__((ext_vector_type(16))) __bf16 v16bf;
typedef __attribute__((ext_vector_type(8)))  float  v8f;

__device__ __forceinline__ unsigned short f2bf(float f) {
  unsigned int u = __float_as_uint(f);
  unsigned int r = u + 0x7FFFu + ((u >> 16) & 1u);   // round-to-nearest-even
  return (unsigned short)(r >> 16);
}
__device__ __forceinline__ float bf2f(unsigned short s) {
  return __uint_as_float(((unsigned int)s) << 16);
}
__device__ __forceinline__ float sigm(float x) { return 1.0f / (1.0f + __expf(-x)); }

union FragBF { v16bf v; uint4 q[2]; unsigned short s[16]; };

// A fragment (16x32 bf16, row-major source, leading dim lda):
// lane: m = m0 + lane%16; elems 0..15 -> K = k0 + (lane/16)*16 + e
__device__ __forceinline__ v16bf load_a_frag(const unsigned short* A, int lda,
                                             int m0, int k0, int lane) {
  int m  = m0 + (lane & 15);
  int kk = k0 + ((lane >> 4) << 4);
  const uint4* p = reinterpret_cast<const uint4*>(A + (size_t)m * lda + kk);
  FragBF f; f.q[0] = p[0]; f.q[1] = p[1];   // +8 elems = +1 uint4
  return f.v;
}

// B fragment: pre-packed, 32 lanes x 16 contiguous bf16 per (kt,nt) tile
__device__ __forceinline__ v16bf load_b_frag(const unsigned short* P, int tileIdx, int lane) {
  const uint4* p = reinterpret_cast<const uint4*>(P + ((size_t)tileIdx * 32 + lane) * 16);
  FragBF f; f.q[0] = p[0]; f.q[1] = p[1];
  return f.v;
}

__device__ __forceinline__ v8f wmma_bf16(v16bf a, v16bf b, v8f c) {
  return __builtin_amdgcn_wmma_f32_16x16x32_bf16(false, a, false, b, (short)0, c, false, false);
}

// ---------------------------------------------------------------------------
// Pack fp32 weight [K][N] row-major into WMMA B-fragment order:
// tileIdx = nt*KT + kt ; per lane (n = nt*16 + lane%16) 16 contiguous K values
// K = kt*32 + (lane/16)*16 + e
__global__ void pack_weight_bf16(const float* __restrict__ W, unsigned short* __restrict__ P,
                                 int K, int N) {
  int idx = blockIdx.x * blockDim.x + threadIdx.x;
  int KT = K >> 5, NT = N >> 4;
  int total = KT * NT * 32;
  if (idx >= total) return;
  int lane = idx & 31;
  int tile = idx >> 5;
  int kt = tile % KT;
  int nt = tile / KT;
  int n = (nt << 4) + (lane & 15);
  int kbase = (kt << 5) + ((lane >> 4) << 4);
  union { unsigned short s[16]; uint4 q[2]; } tmp;
#pragma unroll
  for (int e = 0; e < 16; ++e)
    tmp.s[e] = f2bf(W[(size_t)(kbase + e) * N + n]);
  uint4* dst = reinterpret_cast<uint4*>(P + (size_t)idx * 16);
  dst[0] = tmp.q[0];
  dst[1] = tmp.q[1];
}

__global__ void f32_to_bf16_kernel(const float* __restrict__ src,
                                   unsigned short* __restrict__ dst, size_t n) {
  size_t i = (size_t)blockIdx.x * blockDim.x + threadIdx.x;
  size_t stride = (size_t)gridDim.x * blockDim.x;
  for (; i < n; i += stride) dst[i] = f2bf(src[i]);
}

__global__ void init_state(float* __restrict__ h, float* __restrict__ c,
                           unsigned short* __restrict__ hb, int n) {
  int i = blockIdx.x * blockDim.x + threadIdx.x;
  if (i < n) { h[i] = 0.0f; c[i] = 0.0f; hb[i] = 0; }
}

// ---------------------------------------------------------------------------
// Big input projection: C_bf16[M][N] = A_bf16[M][1024] @ Wpacked + bias
// One wave per 32(M) x 64(N) tile: 2 A-frags x 4 B-frags -> 8 accumulators.
// Double-buffered fragment loads so WMMAs of stage k overlap loads of k+1.
__global__ void __launch_bounds__(32)
gemm_xproj(const unsigned short* __restrict__ A, const unsigned short* __restrict__ BP,
           const float* __restrict__ bias, unsigned short* __restrict__ C, int M, int N) {
  const int K = 1024, KT = 32;
  int lane = threadIdx.x;
  int NS = N >> 6;                 // 64-column strips
  int strip = blockIdx.x;
  int mt = strip / NS;
  int ns = strip - mt * NS;
  int m0 = mt << 5;                // 32 rows
  int nt0 = ns << 2;               // 4 n-tiles
  v8f acc[2][4] = {};
  v16bf abuf[2][2], bbuf[2][4];
#pragma unroll
  for (int i = 0; i < 2; ++i) abuf[0][i] = load_a_frag(A, K, m0 + (i << 4), 0, lane);
#pragma unroll
  for (int j = 0; j < 4; ++j) bbuf[0][j] = load_b_frag(BP, (nt0 + j) * KT, lane);
#pragma unroll
  for (int kt = 0; kt < KT; ++kt) {
    int cur = kt & 1, nxt = cur ^ 1;
    if (kt + 1 < KT) {
#pragma unroll
      for (int i = 0; i < 2; ++i)
        abuf[nxt][i] = load_a_frag(A, K, m0 + (i << 4), (kt + 1) << 5, lane);
#pragma unroll
      for (int j = 0; j < 4; ++j)
        bbuf[nxt][j] = load_b_frag(BP, (nt0 + j) * KT + kt + 1, lane);
    }
    if (kt + 4 < KT)   // prefetch weight stream a few stages ahead
      __builtin_prefetch(BP + ((size_t)((nt0) * KT + kt + 4) * 32 + lane) * 16, 0, 1);
#pragma unroll
    for (int i = 0; i < 2; ++i)
#pragma unroll
      for (int j = 0; j < 4; ++j)
        acc[i][j] = wmma_bf16(abuf[cur][i], bbuf[cur][j], acc[i][j]);
  }
  int nl = lane & 15, mh = (lane >> 4) << 3;
#pragma unroll
  for (int i = 0; i < 2; ++i) {
#pragma unroll
    for (int j = 0; j < 4; ++j) {
      int n = ((nt0 + j) << 4) + nl;
      float bv = bias[n];
#pragma unroll
      for (int r = 0; r < 8; ++r) {
        int m = m0 + (i << 4) + mh + r;
        C[(size_t)m * N + n] = f2bf(acc[i][j][r] + bv);
      }
    }
  }
}

// ---------------------------------------------------------------------------
// Recurrent step GEMMs: 4 waves per block split K (each wave 8 of 32 k-tiles,
// serial WMMA chain of 8), partials combined via LDS, wave 0 runs epilogue.

// Step 1: m = xmi * (h @ Wmh + bmh)   -> mb (bf16 [64][1024])
__global__ void __launch_bounds__(128)
step_m(const unsigned short* __restrict__ hb, const unsigned short* __restrict__ WP,
       const float* __restrict__ bmh, const unsigned short* __restrict__ xpt,
       unsigned short* __restrict__ mb) {
  const int H = 1024, KT = 32, LD5 = 5120;
  int lane = threadIdx.x & 31;
  int w = threadIdx.x >> 5;
  int tile = blockIdx.x;
  int mt = tile & 3, nt = tile >> 2;
  int m0 = mt << 4, base = nt * KT;
  int kt0 = w << 3;
  v8f acc = {};
  v16bf a0 = load_a_frag(hb, H, m0, kt0 << 5, lane);
  v16bf b0 = load_b_frag(WP, base + kt0, lane);
#pragma unroll
  for (int k = 0; k < 8; ++k) {
    v16bf an, bn;
    if (k + 1 < 8) {
      an = load_a_frag(hb, H, m0, (kt0 + k + 1) << 5, lane);
      bn = load_b_frag(WP, base + kt0 + k + 1, lane);
    }
    acc = wmma_bf16(a0, b0, acc);
    if (k + 1 < 8) { a0 = an; b0 = bn; }
  }
  __shared__ float red[3][8][32];
  if (w > 0) {
#pragma unroll
    for (int r = 0; r < 8; ++r) red[w - 1][r][lane] = acc[r];
  }
  __syncthreads();
  if (w == 0) {
#pragma unroll
    for (int r = 0; r < 8; ++r)
      acc[r] += red[0][r][lane] + red[1][r][lane] + red[2][r][lane];
    int n = (nt << 4) + (lane & 15);
    float bv = bmh[n];
    int mb0 = m0 + ((lane >> 4) << 3);
#pragma unroll
    for (int r = 0; r < 8; ++r) {
      int m = mb0 + r;
      float xmi = bf2f(xpt[(size_t)m * LD5 + n]);         // x_mi block (cols 0..H)
      mb[(size_t)m * H + n] = f2bf(xmi * (acc[r] + bv));
    }
  }
}

// Step 2: g = xhi + m @ Whh + bhh    -> g (f32), gb (bf16)
__global__ void __launch_bounds__(128)
step_g(const unsigned short* __restrict__ mbin, const unsigned short* __restrict__ WP,
       const float* __restrict__ bhh, const unsigned short* __restrict__ xpt,
       float* __restrict__ g, unsigned short* __restrict__ gb) {
  const int H = 1024, KT = 32, LD5 = 5120;
  int lane = threadIdx.x & 31;
  int w = threadIdx.x >> 5;
  int tile = blockIdx.x;
  int mt = tile & 3, nt = tile >> 2;
  int m0 = mt << 4, base = nt * KT;
  int kt0 = w << 3;
  v8f acc = {};
  v16bf a0 = load_a_frag(mbin, H, m0, kt0 << 5, lane);
  v16bf b0 = load_b_frag(WP, base + kt0, lane);
#pragma unroll
  for (int k = 0; k < 8; ++k) {
    v16bf an, bn;
    if (k + 1 < 8) {
      an = load_a_frag(mbin, H, m0, (kt0 + k + 1) << 5, lane);
      bn = load_b_frag(WP, base + kt0 + k + 1, lane);
    }
    acc = wmma_bf16(a0, b0, acc);
    if (k + 1 < 8) { a0 = an; b0 = bn; }
  }
  __shared__ float red[3][8][32];
  if (w > 0) {
#pragma unroll
    for (int r = 0; r < 8; ++r) red[w - 1][r][lane] = acc[r];
  }
  __syncthreads();
  if (w == 0) {
#pragma unroll
    for (int r = 0; r < 8; ++r)
      acc[r] += red[0][r][lane] + red[1][r][lane] + red[2][r][lane];
    int n = (nt << 4) + (lane & 15);
    float bv = bhh[n];
    int mb0 = m0 + ((lane >> 4) << 3);
#pragma unroll
    for (int r = 0; r < 8; ++r) {
      int m = mb0 + r;
      float xhi = bf2f(xpt[(size_t)m * LD5 + H + n]);     // x_hi block (cols H..2H)
      float gv = xhi + acc[r] + bv;
      size_t idx = (size_t)m * H + n;
      g[idx] = gv;
      gb[idx] = f2bf(gv);
    }
  }
}

// Step 3: gp = g @ Wg (3 column blocks in-wave), gates, state update, h out
__global__ void __launch_bounds__(128)
step_gates(const unsigned short* __restrict__ gbin, const unsigned short* __restrict__ WgP,
           const float* __restrict__ bg, const unsigned short* __restrict__ xpt,
           const float* __restrict__ g, float* __restrict__ c, float* __restrict__ hf,
           unsigned short* __restrict__ hb, unsigned short* __restrict__ seq_bf,
           float* __restrict__ seq_f32) {
  const int H = 1024, KT = 32, LD5 = 5120;
  int lane = threadIdx.x & 31;
  int w = threadIdx.x >> 5;
  int tile = blockIdx.x;
  int mt = tile & 3, nt = tile >> 2;
  int m0 = mt << 4;
  int bi_base = nt * KT, bo_base = (nt + 64) * KT, bf_base = (nt + 128) * KT;
  int kt0 = w << 3;
  v8f ai = {}, ao = {}, af = {};
  v16bf a0  = load_a_frag(gbin, H, m0, kt0 << 5, lane);
  v16bf vbi = load_b_frag(WgP, bi_base + kt0, lane);
  v16bf vbo = load_b_frag(WgP, bo_base + kt0, lane);
  v16bf vbf = load_b_frag(WgP, bf_base + kt0, lane);
#pragma unroll
  for (int k = 0; k < 8; ++k) {
    v16bf an, bin, bon, bfn;
    if (k + 1 < 8) {
      an  = load_a_frag(gbin, H, m0, (kt0 + k + 1) << 5, lane);
      bin = load_b_frag(WgP, bi_base + kt0 + k + 1, lane);
      bon = load_b_frag(WgP, bo_base + kt0 + k + 1, lane);
      bfn = load_b_frag(WgP, bf_base + kt0 + k + 1, lane);
    }
    ai = wmma_bf16(a0, vbi, ai);
    ao = wmma_bf16(a0, vbo, ao);
    af = wmma_bf16(a0, vbf, af);
    if (k + 1 < 8) { a0 = an; vbi = bin; vbo = bon; vbf = bfn; }
  }
  __shared__ float red[3][3][8][32];
  if (w > 0) {
#pragma unroll
    for (int r = 0; r < 8; ++r) {
      red[w - 1][0][r][lane] = ai[r];
      red[w - 1][1][r][lane] = ao[r];
      red[w - 1][2][r][lane] = af[r];
    }
  }
  __syncthreads();
  if (w == 0) {
#pragma unroll
    for (int r = 0; r < 8; ++r) {
      ai[r] += red[0][0][r][lane] + red[1][0][r][lane] + red[2][0][r][lane];
      ao[r] += red[0][1][r][lane] + red[1][1][r][lane] + red[2][1][r][lane];
      af[r] += red[0][2][r][lane] + red[1][2][r][lane] + red[2][2][r][lane];
    }
    int n = (nt << 4) + (lane & 15);
    float bgi = bg[n], bgo = bg[H + n], bgf = bg[2 * H + n];
    int mb0 = m0 + ((lane >> 4) << 3);
#pragma unroll
    for (int r = 0; r < 8; ++r) {
      int m = mb0 + r;
      size_t xbase = (size_t)m * LD5;
      float iv = sigm(bf2f(xpt[xbase + 2 * H + n]) + ai[r] + bgi);
      float ov = sigm(bf2f(xpt[xbase + 3 * H + n]) + ao[r] + bgo);
      float fv = sigm(bf2f(xpt[xbase + 4 * H + n]) + af[r] + bgf);
      size_t idx = (size_t)m * H + n;
      float gv = g[idx];
      float cv = fv * c[idx] + iv * tanhf(gv);
      c[idx] = cv;
      float hv = tanhf(cv) * ov;
      hf[idx] = hv;
      hb[idx] = f2bf(hv);
      if (seq_bf)  seq_bf[idx] = f2bf(hv);
      if (seq_f32) seq_f32[idx] = hv;
    }
  }
}

// ---------------------------------------------------------------------------
extern "C" void kernel_launch(void* const* d_in, const int* in_sizes, int n_in,
                              void* d_out, int out_size, void* d_ws, size_t ws_size,
                              hipStream_t stream) {
  (void)in_sizes; (void)n_in; (void)out_size; (void)ws_size;
  const int T = 512, Bb = 64, D = 1024, H = 1024, L = 2;
  const int H5 = 5 * H, H3 = 3 * H;
  const size_t Mrows = (size_t)T * Bb;   // 32768

  const float* x   = (const float*)d_in[0];
  const float* Wx  = (const float*)d_in[1];
  const float* bx  = (const float*)d_in[2];
  const float* Wmh = (const float*)d_in[3];
  const float* bmh = (const float*)d_in[4];
  const float* Whh = (const float*)d_in[5];
  const float* bhh = (const float*)d_in[6];
  const float* Wg  = (const float*)d_in[7];
  const float* bg  = (const float*)d_in[8];

  char* ws = (char*)d_ws;
  size_t off = 0;
  auto take = [&](size_t bytes) -> char* {
    char* p = ws + off; off = (off + bytes + 255) & ~(size_t)255; return p;
  };
  unsigned short* xb   = (unsigned short*)take(Mrows * D * 2);
  unsigned short* seqb = (unsigned short*)take(Mrows * H * 2);
  unsigned short* xp   = (unsigned short*)take(Mrows * H5 * 2);
  unsigned short* WxP  = (unsigned short*)take((size_t)L * D * H5 * 2);
  unsigned short* WmhP = (unsigned short*)take((size_t)L * H * H * 2);
  unsigned short* WhhP = (unsigned short*)take((size_t)L * H * H * 2);
  unsigned short* WgP  = (unsigned short*)take((size_t)L * H * H3 * 2);
  float* hf = (float*)take((size_t)Bb * H * 4);
  float* cf = (float*)take((size_t)Bb * H * 4);
  float* gf = (float*)take((size_t)Bb * H * 4);
  unsigned short* hb = (unsigned short*)take((size_t)Bb * H * 2);
  unsigned short* mb = (unsigned short*)take((size_t)Bb * H * 2);
  unsigned short* gb = (unsigned short*)take((size_t)Bb * H * 2);

  // One-time-per-call weight packing (bf16 WMMA fragment order).
  for (int l = 0; l < L; ++l) {
    { int tot = (D >> 5) * (H5 >> 4) * 32;
      pack_weight_bf16<<<(tot + 255) / 256, 256, 0, stream>>>(
          Wx + (size_t)l * D * H5, WxP + (size_t)l * D * H5, D, H5); }
    { int tot = (H >> 5) * (H >> 4) * 32;
      pack_weight_bf16<<<(tot + 255) / 256, 256, 0, stream>>>(
          Wmh + (size_t)l * H * H, WmhP + (size_t)l * H * H, H, H);
      pack_weight_bf16<<<(tot + 255) / 256, 256, 0, stream>>>(
          Whh + (size_t)l * H * H, WhhP + (size_t)l * H * H, H, H); }
    { int tot = (H >> 5) * (H3 >> 4) * 32;
      pack_weight_bf16<<<(tot + 255) / 256, 256, 0, stream>>>(
          Wg + (size_t)l * H * H3, WgP + (size_t)l * H * H3, H, H3); }
  }
  f32_to_bf16_kernel<<<4096, 256, 0, stream>>>(x, xb, Mrows * (size_t)D);
  init_state<<<(Bb * H + 255) / 256, 256, 0, stream>>>(hf, cf, hb, Bb * H);

  float* out_seq = (float*)d_out;
  float* out_h = out_seq + Mrows * H;
  float* out_c = out_h + (size_t)L * Bb * H;

  for (int l = 0; l < L; ++l) {
    const unsigned short* Ain = (l == 0) ? xb : seqb;
    int nblocks = (int)(Mrows / 32) * (H5 / 64);
    gemm_xproj<<<nblocks, 32, 0, stream>>>(Ain, WxP + (size_t)l * D * H5,
                                           bx + (size_t)l * H5, xp, (int)Mrows, H5);
    for (int t = 0; t < T; ++t) {
      const unsigned short* xpt = xp + (size_t)t * Bb * H5;
      step_m<<<256, 128, 0, stream>>>(hb, WmhP + (size_t)l * H * H,
                                      bmh + (size_t)l * H, xpt, mb);
      step_g<<<256, 128, 0, stream>>>(mb, WhhP + (size_t)l * H * H,
                                      bhh + (size_t)l * H, xpt, gf, gb);
      unsigned short* sbf = (l == 0) ? (seqb + (size_t)t * Bb * H) : nullptr;
      float* sf32 = (l == 1) ? (out_seq + (size_t)t * Bb * H) : nullptr;
      step_gates<<<256, 128, 0, stream>>>(gb, WgP + (size_t)l * H * H3,
                                          bg + (size_t)l * H3, xpt, gf, cf, hf, hb,
                                          sbf, sf32);
    }
    hipMemcpyAsync(out_h + (size_t)l * Bb * H, hf, (size_t)Bb * H * 4,
                   hipMemcpyDeviceToDevice, stream);
    hipMemcpyAsync(out_c + (size_t)l * Bb * H, cf, (size_t)Bb * H * 4,
                   hipMemcpyDeviceToDevice, stream);
  }
}
